// HiMAN_12524124635865
// MI455X (gfx1250) — compile-verified, matrix operand order checked by reading, not run on hardware
//
#include <hip/hip_runtime.h>
#include <hip/hip_bf16.h>
#include <math.h>

// ---------------------------------------------------------------------------
// HiMAN on gfx1250: bf16 WMMA GEMMs (v_wmma_f32_16x16x32_bf16) for all large
// matmuls, fused attention-score epilogue, wave32 shfl reductions elsewhere.
// GEMM: 32x128 block tile, 8 waves x (2 WMMA sharing one B frag), K-panel 64
// staged in LDS -> 4 WMMA per barrier pair, b128 staging, B-panel prefetch.
// ---------------------------------------------------------------------------

typedef __bf16 bf16_t;
typedef bf16_t bf16x16 __attribute__((ext_vector_type(16)));
typedef float  f32x8   __attribute__((ext_vector_type(8)));

#define NEGV  (-1.0e9f)
#define NINF  (-1.0e30f)

// model dims
#define BN     32
#define SLN    24
#define WLN    64
#define EMBN   300
#define EMBP   320     // padded K (multiple of 64)
#define HN     512
#define H3N    1536
#define D2N    1024
#define INN    8
#define IGN    1024
#define VOCABN 30000
#define NCLSN  3
#define NSEQ   (BN*SLN)    // 768
#define NTOK   (NSEQ*WLN)  // 49152

struct __align__(16) bf16x8 { bf16_t v[8]; };

__device__ __forceinline__ float sigf(float x) { return 1.f / (1.f + __expf(-x)); }

// ---------------------------------------------------------------------------
// Generic bf16 GEMM: C(MxN) = A(MxK) * W(NxK)^T [+ bias]
// block = 256 threads = 8 waves; tile 32(M) x 128(N); K staged in 64-panels.
// Each wave owns a unique 16-wide N strip (no duplicated B fetches) and
// computes two 16x16 C tiles (M 0-15 / 16-31) sharing each B fragment.
// EPI==0: optional bias, store f32 and/or bf16.
// EPI==1: val = relu(acc+bias)*w2; row-reduce across lanes; atomicAdd into red.
// Requires M%32==0, N%128==0, K%64==0 (true for every call site here).
// ---------------------------------------------------------------------------
template <int EPI>
__global__ __launch_bounds__(256) void himan_gemm_k(
    const bf16_t* __restrict__ A, int lda,
    const bf16_t* __restrict__ W, int ldw,
    const float* __restrict__ bias,
    float* __restrict__ C, bf16_t* __restrict__ Cbf, int ldc,
    const float* __restrict__ w2, float* __restrict__ red,
    int M, int N, int K)
{
    __shared__ bf16_t As[32][72];   // 32 x 64 K-panel + pad
    const int tid  = threadIdx.x;
    const int wave = tid >> 5;      // 0..7 -> N strip
    const int lane = tid & 31;
    const int tileM = blockIdx.y * 32;
    const int tileN = blockIdx.x * 128;
    const bool hi  = lane >= 16;
    const int lrow = lane & 15;

    const int ldsRow = tid >> 3;          // 0..31
    const int ldsCol = (tid & 7) * 8;     // 0..56
    const bf16_t* agrow = A + (size_t)(tileM + ldsRow) * lda + ldsCol;
    const bf16_t* wrow  = W + (size_t)(tileN + wave * 16 + lrow) * ldw;

    f32x8 acc0 = {0.f, 0.f, 0.f, 0.f, 0.f, 0.f, 0.f, 0.f};
    f32x8 acc1 = {0.f, 0.f, 0.f, 0.f, 0.f, 0.f, 0.f, 0.f};

    for (int k0 = 0; k0 < K; k0 += 64) {
        // stage 32x64 A panel: one b128 global load + one b128 ds store/thread
        *reinterpret_cast<bf16x8*>(&As[ldsRow][ldsCol]) =
            *reinterpret_cast<const bf16x8*>(agrow + k0);
        // prefetch next B panel into cache while we compute this one
        if (k0 + 64 < K) __builtin_prefetch(wrow + k0 + 64, 0, 1);
        __syncthreads();

#pragma unroll
        for (int ks = 0; ks < 2; ++ks) {
            const int kb = ks * 32;
            bf16x16 a0, a1, bv;
            const bf16_t* ar0 = &As[lrow][kb];
            const bf16_t* ar1 = &As[16 + lrow][kb];
            const bf16_t* br  = wrow + k0 + kb;
#pragma unroll
            for (int v = 0; v < 8; ++v) {
                // 16-bit A/B fragment layout (ISA 7.12.2): halves 2v,2v+1 hold
                // K = 2v (+8 if v>=4) (+8 if lane>=16), K+1
                const int kk = 2 * v + (v >= 4 ? 8 : 0) + (hi ? 8 : 0);
                a0[2 * v] = ar0[kk]; a0[2 * v + 1] = ar0[kk + 1];
                a1[2 * v] = ar1[kk]; a1[2 * v + 1] = ar1[kk + 1];
                bv[2 * v] = br[kk];  bv[2 * v + 1] = br[kk + 1];
            }
            acc0 = __builtin_amdgcn_wmma_f32_16x16x32_bf16(
                false, a0, false, bv, (short)0, acc0, false, false);
            acc1 = __builtin_amdgcn_wmma_f32_16x16x32_bf16(
                false, a1, false, bv, (short)0, acc1, false, false);
        }
        __syncthreads();
    }

    const int cRow = tileM + (hi ? 8 : 0);       // tile0 rows; tile1 = +16
    const int cCol = tileN + wave * 16 + lrow;
    if (EPI == 0) {
        const float bv = bias ? bias[cCol] : 0.f;
#pragma unroll
        for (int r = 0; r < 8; ++r) {
            const float v0 = acc0[r] + bv;
            const float v1 = acc1[r] + bv;
            if (C) {
                C[(size_t)(cRow + r) * ldc + cCol]      = v0;
                C[(size_t)(cRow + 16 + r) * ldc + cCol] = v1;
            }
            if (Cbf) {
                Cbf[(size_t)(cRow + r) * ldc + cCol]      = (bf16_t)v0;
                Cbf[(size_t)(cRow + 16 + r) * ldc + cCol] = (bf16_t)v1;
            }
        }
    } else {
        const float bv  = bias[cCol];
        const float w2v = w2[cCol];
#pragma unroll
        for (int r = 0; r < 8; ++r) {
            float v0 = fmaxf(acc0[r] + bv, 0.f) * w2v;
            float v1 = fmaxf(acc1[r] + bv, 0.f) * w2v;
            v0 += __shfl_xor(v0, 1, 32);
            v0 += __shfl_xor(v0, 2, 32);
            v0 += __shfl_xor(v0, 4, 32);
            v0 += __shfl_xor(v0, 8, 32);
            v1 += __shfl_xor(v1, 1, 32);
            v1 += __shfl_xor(v1, 2, 32);
            v1 += __shfl_xor(v1, 4, 32);
            v1 += __shfl_xor(v1, 8, 32);
            if (lrow == 0) {
                atomicAdd(&red[cRow + r], v0);
                atomicAdd(&red[cRow + 16 + r], v1);
            }
        }
    }
}

// f32 -> bf16 convert with optional column offset / zero K-padding
__global__ void himan_cvt_k(const float* __restrict__ src, bf16_t* __restrict__ dst,
                            int rows, int srcld, int srcoff, int cols, int dstld)
{
    const int i = blockIdx.x * blockDim.x + threadIdx.x;
    if (i >= rows * dstld) return;
    const int r = i / dstld, c = i - r * dstld;
    const float v = (c < cols) ? src[(size_t)r * srcld + srcoff + c] : 0.f;
    dst[i] = (bf16_t)v;
}

// embedding gather into padded bf16 rows
__global__ void himan_gather_k(const int* __restrict__ words, const bf16_t* __restrict__ embp,
                               bf16_t* __restrict__ tok, int n, int kp)
{
    const int i = blockIdx.x * blockDim.x + threadIdx.x;
    if (i >= n * kp) return;
    const int t = i / kp, c = i - t * kp;
    tok[i] = embp[(size_t)words[t] * kp + c];
}

__global__ void himan_fill_k(float* __restrict__ dst, const float* __restrict__ v, int n)
{
    const int i = blockIdx.x * blockDim.x + threadIdx.x;
    if (i < n) dst[i] = v[0];
}

// GRU gates: r=sig(xr+hr), z=sig(xz+hz), n=tanh(xn+r*hn), h=(1-z)n+z*h
__global__ void himan_gru_gate_k(const float* __restrict__ gh,   // (batch,3H) incl bhh
                                 const float* __restrict__ xp,   // (batch*seqLen,3H) incl bih
                                 float* __restrict__ h, bf16_t* __restrict__ hbf,
                                 bf16_t* __restrict__ out_bf, float* __restrict__ out_f,
                                 int batch, int seqLen, int t, int dirOff)
{
    const int i = blockIdx.x * blockDim.x + threadIdx.x;
    if (i >= batch * HN) return;
    const int b = i / HN, d = i - b * HN;
    const float* g = gh + (size_t)b * H3N;
    const float* x = xp + ((size_t)b * seqLen + t) * H3N;
    const float r  = sigf(x[d] + g[d]);
    const float z  = sigf(x[HN + d] + g[HN + d]);
    const float nn = tanhf(x[2 * HN + d] + r * g[2 * HN + d]);
    const float hv = (1.f - z) * nn + z * h[i];
    h[i]   = hv;
    hbf[i] = (bf16_t)hv;
    const size_t o = ((size_t)b * seqLen + t) * (2 * HN) + dirOff + d;
    out_bf[o] = (bf16_t)hv;
    if (out_f) out_f[o] = hv;
}

// word-level attflat: masked softmax over WL + weighted pool over ow
__global__ __launch_bounds__(256) void himan_word_pool_k(
    const float* __restrict__ a2, const int* __restrict__ words,
    const bf16_t* __restrict__ ow, bf16_t* __restrict__ pooled)
{
    __shared__ float sh[WLN];
    __shared__ float inv_s;
    const int s = blockIdx.x;
    const int tid = threadIdx.x;
    if (tid < WLN) {
        const int tok = words[s * WLN + tid];
        sh[tid] = (tok == 0) ? NEGV : a2[s * WLN + tid];
    }
    __syncthreads();
    if (tid == 0) {
        float mx = sh[0];
        for (int t = 1; t < WLN; ++t) mx = fmaxf(mx, sh[t]);
        float sum = 0.f;
        for (int t = 0; t < WLN; ++t) { const float e = __expf(sh[t] - mx); sh[t] = e; sum += e; }
        inv_s = 1.f / sum;
    }
    __syncthreads();
    const float inv = inv_s;
    for (int d = tid; d < D2N; d += 256) {
        float acc = 0.f;
        const bf16_t* op = ow + (size_t)s * WLN * D2N + d;
        for (int t = 0; t < WLN; ++t) acc += sh[t] * (float)op[(size_t)t * D2N];
        pooled[(size_t)s * D2N + d] = (bf16_t)(acc * inv);
    }
}

// sentence attention weights: masked softmax over SL per batch (32 lanes)
__global__ void himan_sen_softmax_k(const float* __restrict__ a2s,
                                    const int* __restrict__ sen_num,
                                    float* __restrict__ sen_a)
{
    const int b = blockIdx.x;
    const int lane = threadIdx.x;
    const int sn = sen_num[b];
    const float v = (lane < SLN) ? ((lane < sn) ? a2s[b * SLN + lane] : NEGV) : NINF;
    float m = v;
    for (int o = 16; o; o >>= 1) m = fmaxf(m, __shfl_xor(m, o, 32));
    const float e = __expf(v - m);
    float sum = e;
    for (int o = 16; o; o >>= 1) sum += __shfl_xor(sum, o, 32);
    if (lane < SLN) sen_a[b * SLN + lane] = e / sum;
}

__global__ void himan_cat_k(const float* __restrict__ g, const float* __restrict__ s,
                            bf16_t* __restrict__ dst, int rows, int ig)
{
    const int i = blockIdx.x * blockDim.x + threadIdx.x;
    if (i >= rows * 2 * ig) return;
    const int r = i / (2 * ig), c = i - r * 2 * ig;
    const float v = (c < ig) ? g[(size_t)r * ig + c] : s[(size_t)r * ig + (c - ig)];
    dst[i] = (bf16_t)v;
}

// sem[b,i,s] = sum_d h1w[d]*(c*os + (1-c)*img), c = sig(sent_part+img_part)
__global__ __launch_bounds__(256) void himan_sem_k(
    const float* __restrict__ sent_part, const float* __restrict__ img_part,
    const float* __restrict__ osf, const float* __restrict__ imgf,
    const float* __restrict__ h1w, float* __restrict__ sem)
{
    const int w = (blockIdx.x * blockDim.x + threadIdx.x) >> 5;
    const int lane = threadIdx.x & 31;
    if (w >= BN * INN * SLN) return;
    const int s = w % SLN;
    const int i = (w / SLN) % INN;
    const int b = w / (SLN * INN);
    const float* sp = sent_part + ((size_t)b * SLN + s) * D2N;
    const float* os = osf       + ((size_t)b * SLN + s) * D2N;
    const float* ip = img_part  + ((size_t)b * INN + i) * D2N;
    const float* im = imgf      + ((size_t)b * INN + i) * D2N;
    float acc = 0.f;
    for (int d = lane; d < D2N; d += 32) {
        const float c = sigf(sp[d] + ip[d]);
        acc += (c * os[d] + (1.f - c) * im[d]) * h1w[d];
    }
    for (int o = 16; o; o >>= 1) acc += __shfl_xor(acc, o, 32);
    if (lane == 0) sem[w] = acc;
}

// total = 0.3*softmax(pos) + 0.7*softmax(sem) + sen_a ; one wave per (b,i)
__global__ __launch_bounds__(256) void himan_total_k(
    const float* __restrict__ sem, const float* __restrict__ dis,
    const float* __restrict__ sen_a, const int* __restrict__ sen_num,
    float* __restrict__ tot)
{
    const int w = (blockIdx.x * blockDim.x + threadIdx.x) >> 5;
    const int lane = threadIdx.x & 31;
    if (w >= BN * INN) return;
    const int b = w / INN;
    const int sn = sen_num[b];

    float sv = (lane < SLN) ? ((lane < sn) ? sem[(size_t)w * SLN + lane] : NEGV) : NINF;
    float m = sv;
    for (int o = 16; o; o >>= 1) m = fmaxf(m, __shfl_xor(m, o, 32));
    const float e = __expf(sv - m);
    float su = e;
    for (int o = 16; o; o >>= 1) su += __shfl_xor(su, o, 32);
    const float smx = e / su;

    const float dv = (lane < SLN) ? dis[(size_t)w * SLN + lane] : 0.f;
    float pv = (lane < SLN) ? ((fabsf(dv) == 0.f) ? NEGV : (40.f - dv)) : NINF;
    float m2 = pv;
    for (int o = 16; o; o >>= 1) m2 = fmaxf(m2, __shfl_xor(m2, o, 32));
    const float e2 = __expf(pv - m2);
    float su2 = e2;
    for (int o = 16; o; o >>= 1) su2 += __shfl_xor(su2, o, 32);
    const float psx = e2 / su2;

    if (lane < SLN)
        tot[(size_t)w * SLN + lane] = 0.3f * psx + 0.7f * smx + sen_a[b * SLN + lane];
}

// x_atted[b,i,d] = sum_s total[b,i,s]*os[b,s,d] + 0.5*img_feat[b,i,d]
__global__ void himan_xatt_k(const float* __restrict__ tot, const float* __restrict__ osf,
                             const float* __restrict__ imgf, float* __restrict__ xatt,
                             bf16_t* __restrict__ xatt_bf)
{
    const int idx = blockIdx.x * blockDim.x + threadIdx.x;
    if (idx >= BN * INN * D2N) return;
    const int d = idx % D2N;
    const int i = (idx / D2N) % INN;
    const int b = idx / (D2N * INN);
    const float* tp = tot + ((size_t)b * INN + i) * SLN;
    float acc = 0.5f * imgf[idx];
    for (int s = 0; s < SLN; ++s) acc += tp[s] * osf[((size_t)b * SLN + s) * D2N + d];
    xatt[idx]    = acc;
    xatt_bf[idx] = (bf16_t)acc;
}

// final attflat: masked softmax over IN + pool over x_atted
__global__ __launch_bounds__(256) void himan_final_pool_k(
    const float* __restrict__ a2f, const float* __restrict__ dis,
    const float* __restrict__ xatt, bf16_t* __restrict__ pooled)
{
    __shared__ float p[INN];
    const int b = blockIdx.x;
    const int tid = threadIdx.x;
    if (tid < INN) {
        float s = 0.f;
        for (int t = 0; t < SLN; ++t) s += fabsf(dis[((size_t)b * INN + tid) * SLN + t]);
        p[tid] = (s == 0.f) ? NEGV : a2f[b * INN + tid];
    }
    __syncthreads();
    if (tid == 0) {
        float mx = p[0];
        for (int i = 1; i < INN; ++i) mx = fmaxf(mx, p[i]);
        float su = 0.f;
        for (int i = 0; i < INN; ++i) { const float e = __expf(p[i] - mx); p[i] = e; su += e; }
        for (int i = 0; i < INN; ++i) p[i] /= su;
    }
    __syncthreads();
    for (int d = tid; d < D2N; d += 256) {
        float acc = 0.f;
        for (int i = 0; i < INN; ++i) acc += p[i] * xatt[((size_t)b * INN + i) * D2N + d];
        pooled[(size_t)b * D2N + d] = (bf16_t)acc;
    }
}

// out[b,c] = sv[b]·fin_W[c] + fin_b[c] ; one wave per (b,c)
__global__ void himan_final_out_k(const float* __restrict__ sv, const float* __restrict__ finW,
                                  const float* __restrict__ finb, float* __restrict__ out)
{
    const int w = (blockIdx.x * blockDim.x + threadIdx.x) >> 5;
    const int lane = threadIdx.x & 31;
    if (w >= BN * NCLSN) return;
    const int c = w % NCLSN, b = w / NCLSN;
    float acc = 0.f;
    for (int d = lane; d < D2N; d += 32)
        acc += sv[(size_t)b * D2N + d] * finW[(size_t)c * D2N + d];
    for (int o = 16; o; o >>= 1) acc += __shfl_xor(acc, o, 32);
    if (lane == 0) out[b * NCLSN + c] = acc + finb[c];
}

// ---------------------------------------------------------------------------
// host-side launch helpers
// ---------------------------------------------------------------------------
static void launch_gemm(hipStream_t st, const bf16_t* A, int lda, const bf16_t* W, int ldw,
                        const float* bias, float* C, bf16_t* Cbf, int ldc,
                        int M, int N, int K)
{
    dim3 g(N / 128, M / 32);
    himan_gemm_k<0><<<g, 256, 0, st>>>(A, lda, W, ldw, bias, C, Cbf, ldc,
                                       nullptr, nullptr, M, N, K);
}
static void launch_gemm_att(hipStream_t st, const bf16_t* A, int lda, const bf16_t* W, int ldw,
                            const float* bias, const float* w2, float* red,
                            int M, int N, int K)
{
    dim3 g(N / 128, M / 32);
    himan_gemm_k<1><<<g, 256, 0, st>>>(A, lda, W, ldw, bias, nullptr, nullptr, N,
                                       w2, red, M, N, K);
}

extern "C" void kernel_launch(void* const* d_in, const int* in_sizes, int n_in,
                              void* d_out, int out_size, void* d_ws, size_t ws_size,
                              hipStream_t stream)
{
    (void)in_sizes; (void)n_in; (void)out_size; (void)ws_size;

    const int*   x_t       = (const int*)  d_in[0];
    const int*   sen_num   = (const int*)  d_in[1];
    /* d_in[2] sen_len unused by reference */
    const float* x_v_g     = (const float*)d_in[3];
    const float* x_v_s     = (const float*)d_in[4];
    /* d_in[5], d_in[6] unused by reference */
    const float* x_img_dis = (const float*)d_in[7];
    const float* emb       = (const float*)d_in[8];
    const float* wg_Wih    = (const float*)d_in[9];
    const float* wg_Whh    = (const float*)d_in[10];
    const float* wg_bih    = (const float*)d_in[11];
    const float* wg_bhh    = (const float*)d_in[12];
    const float* waf_W1    = (const float*)d_in[13];
    const float* waf_b1    = (const float*)d_in[14];
    const float* waf_W2    = (const float*)d_in[15];
    const float* waf_b2    = (const float*)d_in[16];
    const float* waf_Wm    = (const float*)d_in[17];
    const float* waf_bm    = (const float*)d_in[18];
    const float* sg_Wih    = (const float*)d_in[19];
    const float* sg_Whh    = (const float*)d_in[20];
    const float* sg_bih    = (const float*)d_in[21];
    const float* sg_bhh    = (const float*)d_in[22];
    const float* saf_W1    = (const float*)d_in[23];
    const float* saf_b1    = (const float*)d_in[24];
    const float* saf_W2    = (const float*)d_in[25];
    const float* saf_b2    = (const float*)d_in[26];
    const float* saf_Wm    = (const float*)d_in[27];
    const float* saf_bm    = (const float*)d_in[28];
    const float* h1_W      = (const float*)d_in[29];
    const float* img_W     = (const float*)d_in[30];
    const float* img_b     = (const float*)d_in[31];
    const float* cc_W      = (const float*)d_in[32];
    const float* cc_b      = (const float*)d_in[33];
    const float* fin_W     = (const float*)d_in[34];
    const float* fin_b     = (const float*)d_in[35];
    float* out = (float*)d_out;

    // ---- workspace layout (256B aligned slices) ----
    char* base = (char*)d_ws;
    size_t off = 0;
    auto alloc = [&](size_t bytes) -> char* {
        char* p = base + off;
        off += (bytes + 255) & ~(size_t)255;
        return p;
    };
    bf16_t* emb_bf    = (bf16_t*)alloc((size_t)VOCABN * EMBP * 2);
    bf16_t* wgWih_bf  = (bf16_t*)alloc((size_t)2 * H3N * EMBP * 2);
    bf16_t* wgWhh_bf  = (bf16_t*)alloc((size_t)2 * H3N * HN * 2);
    bf16_t* wafW1_bf  = (bf16_t*)alloc((size_t)D2N * D2N * 2);
    bf16_t* wafWm_bf  = (bf16_t*)alloc((size_t)D2N * D2N * 2);
    bf16_t* sgWih_bf  = (bf16_t*)alloc((size_t)2 * H3N * D2N * 2);
    bf16_t* sgWhh_bf  = (bf16_t*)alloc((size_t)2 * H3N * HN * 2);
    bf16_t* safW1_bf  = (bf16_t*)alloc((size_t)D2N * D2N * 2);
    bf16_t* safWm_bf  = (bf16_t*)alloc((size_t)D2N * D2N * 2);
    bf16_t* imgW_bf   = (bf16_t*)alloc((size_t)D2N * 2 * IGN * 2);
    bf16_t* Ws_bf     = (bf16_t*)alloc((size_t)D2N * D2N * 2);
    bf16_t* Wi_bf     = (bf16_t*)alloc((size_t)D2N * D2N * 2);

    bf16_t* tok_bf    = (bf16_t*)alloc((size_t)NTOK * EMBP * 2);
    float*  xp        = (float*) alloc((size_t)NTOK * H3N * 4);   // reused per dir + stage
    float*  h         = (float*) alloc((size_t)NSEQ * HN * 4);
    bf16_t* h_bf      = (bf16_t*)alloc((size_t)NSEQ * HN * 2);
    float*  gh        = (float*) alloc((size_t)NSEQ * H3N * 4);
    bf16_t* ow_bf     = (bf16_t*)alloc((size_t)NTOK * D2N * 2);
    float*  a2        = (float*) alloc((size_t)NTOK * 4);
    bf16_t* pooled_bf = (bf16_t*)alloc((size_t)NSEQ * D2N * 2);
    bf16_t* s_bf      = (bf16_t*)alloc((size_t)NSEQ * D2N * 2);
    float*  os_f      = (float*) alloc((size_t)NSEQ * D2N * 4);
    bf16_t* os_bf     = (bf16_t*)alloc((size_t)NSEQ * D2N * 2);
    float*  a2s       = (float*) alloc((size_t)NSEQ * 4);
    float*  sen_a     = (float*) alloc((size_t)NSEQ * 4);
    bf16_t* cat_bf    = (bf16_t*)alloc((size_t)BN * INN * 2 * IGN * 2);
    float*  img_feat  = (float*) alloc((size_t)BN * INN * D2N * 4);
    bf16_t* imgf_bf   = (bf16_t*)alloc((size_t)BN * INN * D2N * 2);
    float*  sent_part = (float*) alloc((size_t)NSEQ * D2N * 4);
    float*  img_part  = (float*) alloc((size_t)BN * INN * D2N * 4);
    float*  sem       = (float*) alloc((size_t)BN * INN * SLN * 4);
    float*  tot       = (float*) alloc((size_t)BN * INN * SLN * 4);
    float*  xatt      = (float*) alloc((size_t)BN * INN * D2N * 4);
    bf16_t* xatt_bf   = (bf16_t*)alloc((size_t)BN * INN * D2N * 2);
    float*  a2f       = (float*) alloc((size_t)BN * INN * 4);
    bf16_t* poolf_bf  = (bf16_t*)alloc((size_t)BN * D2N * 2);
    float*  sv        = (float*) alloc((size_t)BN * D2N * 4);

    // ---- weight / embedding conversion to bf16 (K padded where needed) ----
    auto cvt = [&](const float* src, bf16_t* dst, int rows, int srcld, int srcoff,
                   int cols, int dstld) {
        const int total = rows * dstld;
        himan_cvt_k<<<(total + 255) / 256, 256, 0, stream>>>(src, dst, rows, srcld,
                                                             srcoff, cols, dstld);
    };
    cvt(emb,    emb_bf,   VOCABN,  EMBN,    0,    EMBN,    EMBP);
    cvt(wg_Wih, wgWih_bf, 2 * H3N, EMBN,    0,    EMBN,    EMBP);
    cvt(wg_Whh, wgWhh_bf, 2 * H3N, HN,      0,    HN,      HN);
    cvt(waf_W1, wafW1_bf, D2N,     D2N,     0,    D2N,     D2N);
    cvt(waf_Wm, wafWm_bf, D2N,     D2N,     0,    D2N,     D2N);
    cvt(sg_Wih, sgWih_bf, 2 * H3N, D2N,     0,    D2N,     D2N);
    cvt(sg_Whh, sgWhh_bf, 2 * H3N, HN,      0,    HN,      HN);
    cvt(saf_W1, safW1_bf, D2N,     D2N,     0,    D2N,     D2N);
    cvt(saf_Wm, safWm_bf, D2N,     D2N,     0,    D2N,     D2N);
    cvt(img_W,  imgW_bf,  D2N,     2 * IGN, 0,    2 * IGN, 2 * IGN);
    cvt(cc_W,   Ws_bf,    D2N,     2 * D2N, 0,    D2N,     D2N);
    cvt(cc_W,   Wi_bf,    D2N,     2 * D2N, D2N,  D2N,     D2N);

    // ---- embedding gather ----
    himan_gather_k<<<((NTOK * EMBP) + 255) / 256, 256, 0, stream>>>(
        x_t, emb_bf, tok_bf, NTOK, EMBP);

    // ---- word-level BiGRU ----
    for (int dir = 0; dir < 2; ++dir) {
        // xp = tok * Wih^T + bih  (one big GEMM, M=49152 N=1536 K=320)
        launch_gemm(stream, tok_bf, EMBP, wgWih_bf + (size_t)dir * H3N * EMBP, EMBP,
                    wg_bih + dir * H3N, xp, nullptr, H3N, NTOK, H3N, EMBP);
        hipMemsetAsync(h,    0, (size_t)NSEQ * HN * 4, stream);
        hipMemsetAsync(h_bf, 0, (size_t)NSEQ * HN * 2, stream);
        for (int stp = 0; stp < WLN; ++stp) {
            const int t = dir ? (WLN - 1 - stp) : stp;
            launch_gemm(stream, h_bf, HN, wgWhh_bf + (size_t)dir * H3N * HN, HN,
                        wg_bhh + dir * H3N, gh, nullptr, H3N, NSEQ, H3N, HN);
            himan_gru_gate_k<<<(NSEQ * HN + 255) / 256, 256, 0, stream>>>(
                gh, xp, h, h_bf, ow_bf, nullptr, NSEQ, WLN, t, dir * HN);
        }
    }

    // ---- word attflat: a2 = relu(ow@W1^T+b1)@W2^T + b2 (fused), softmax+pool, Wm ----
    himan_fill_k<<<(NTOK + 255) / 256, 256, 0, stream>>>(a2, waf_b2, NTOK);
    launch_gemm_att(stream, ow_bf, D2N, wafW1_bf, D2N, waf_b1, waf_W2, a2,
                    NTOK, D2N, D2N);
    himan_word_pool_k<<<NSEQ, 256, 0, stream>>>(a2, x_t, ow_bf, pooled_bf);
    launch_gemm(stream, pooled_bf, D2N, wafWm_bf, D2N, waf_bm, nullptr, s_bf, D2N,
                NSEQ, D2N, D2N);

    // ---- sentence BiGRU (xp buffer reused) ----
    float* xp2 = xp;
    for (int dir = 0; dir < 2; ++dir)
        launch_gemm(stream, s_bf, D2N, sgWih_bf + (size_t)dir * H3N * D2N, D2N,
                    sg_bih + dir * H3N, xp2 + (size_t)dir * NSEQ * H3N, nullptr, H3N,
                    NSEQ, H3N, D2N);
    for (int dir = 0; dir < 2; ++dir) {
        hipMemsetAsync(h,    0, (size_t)BN * HN * 4, stream);
        hipMemsetAsync(h_bf, 0, (size_t)BN * HN * 2, stream);
        for (int stp = 0; stp < SLN; ++stp) {
            const int t = dir ? (SLN - 1 - stp) : stp;
            launch_gemm(stream, h_bf, HN, sgWhh_bf + (size_t)dir * H3N * HN, HN,
                        sg_bhh + dir * H3N, gh, nullptr, H3N, BN, H3N, HN);
            himan_gru_gate_k<<<(BN * HN + 255) / 256, 256, 0, stream>>>(
                gh, xp2 + (size_t)dir * NSEQ * H3N, h, h_bf, os_bf, os_f, BN, SLN, t,
                dir * HN);
        }
    }

    // ---- sentence attention weights (sen_a) ----
    himan_fill_k<<<(NSEQ + 255) / 256, 256, 0, stream>>>(a2s, saf_b2, NSEQ);
    launch_gemm_att(stream, os_bf, D2N, safW1_bf, D2N, saf_b1, saf_W2, a2s,
                    NSEQ, D2N, D2N);
    himan_sen_softmax_k<<<BN, 32, 0, stream>>>(a2s, sen_num, sen_a);

    // ---- image features + gated fusion ----
    himan_cat_k<<<((BN * INN * 2 * IGN) + 255) / 256, 256, 0, stream>>>(
        x_v_g, x_v_s, cat_bf, BN * INN, IGN);
    launch_gemm(stream, cat_bf, 2 * IGN, imgW_bf, 2 * IGN, img_b, img_feat, imgf_bf,
                D2N, BN * INN, D2N, 2 * IGN);
    launch_gemm(stream, os_bf, D2N, Ws_bf, D2N, nullptr, sent_part, nullptr, D2N,
                NSEQ, D2N, D2N);
    launch_gemm(stream, imgf_bf, D2N, Wi_bf, D2N, cc_b, img_part, nullptr, D2N,
                BN * INN, D2N, D2N);
    himan_sem_k<<<((BN * INN * SLN * 32) + 255) / 256, 256, 0, stream>>>(
        sent_part, img_part, os_f, img_feat, h1_W, sem);
    himan_total_k<<<((BN * INN * 32) + 255) / 256, 256, 0, stream>>>(
        sem, x_img_dis, sen_a, sen_num, tot);
    himan_xatt_k<<<((BN * INN * D2N) + 255) / 256, 256, 0, stream>>>(
        tot, os_f, img_feat, xatt, xatt_bf);

    // ---- final attflat + classifier ----
    himan_fill_k<<<((BN * INN) + 255) / 256, 256, 0, stream>>>(a2f, saf_b2, BN * INN);
    launch_gemm_att(stream, xatt_bf, D2N, safW1_bf, D2N, saf_b1, saf_W2, a2f,
                    BN * INN, D2N, D2N);
    himan_final_pool_k<<<BN, 256, 0, stream>>>(a2f, x_img_dis, xatt, poolf_bf);
    launch_gemm(stream, poolf_bf, D2N, safWm_bf, D2N, saf_bm, sv, nullptr, D2N,
                BN, D2N, D2N);
    himan_final_out_k<<<((BN * NCLSN * 32) + 255) / 256, 256, 0, stream>>>(
        sv, fin_W, fin_b, out);
}
